// ReuseEncoderLayer_37383395344627
// MI455X (gfx1250) — matrix operand order repaired and checked by reading, not compiled, not run on hardware
//
#include <hip/hip_runtime.h>
#include <hip/hip_bf16.h>
#include <stdint.h>

// ---------------------------------------------------------------- constants
#define Bb   32
#define Nn   257
#define DIMd 1024
#define Hh   16
#define HDd  64
#define MLPm 4096
#define M_REAL   (Bb * Nn)          // 8224
#define M_PAD    8320               // 65 * 128
#define QP_PAD   320                // padded sequence length (10 * 32)
#define SCALEq   0.125f             // 64^-0.5

typedef __bf16 bf16_t;
typedef __attribute__((ext_vector_type(16))) __bf16 v16bf;
typedef __attribute__((ext_vector_type(8)))  __bf16 v8bf;
typedef __attribute__((ext_vector_type(8)))  float  v8f;

__device__ inline bf16_t f2bf(float f) {
    union { float f; uint32_t u; } v; v.f = f;
    uint32_t r = v.u + 0x7FFFu + ((v.u >> 16) & 1u);   // round-to-nearest-even
    union { uint16_t s; bf16_t b; } o; o.s = (uint16_t)(r >> 16);
    return o.b;
}

__device__ inline v8f wmma_bf16(v16bf a, v16bf b, v8f c) {
    // D(f32 16x16) = A(16x32 bf16) * B(32x16 bf16) + C
    return __builtin_amdgcn_wmma_f32_16x16x32_bf16(
        /*neg_a=*/false, a, /*neg_b=*/false, b,
        /*c_mod=*/(short)0, c, /*reuse_a=*/false, /*reuse_b=*/false);
}

// A-fragment: 16 contiguous halves at [k .. k+15]
__device__ inline v16bf load_afrag(const bf16_t* p) {
    v8bf lo = *(const v8bf*)(p);
    v8bf hi = *(const v8bf*)(p + 8);
    return __builtin_shufflevector(lo, hi, 0,1,2,3,4,5,6,7,8,9,10,11,12,13,14,15);
}

// ---------------- CDNA5 async LDS fill (ASYNCcnt path, ISA 08_async_tensor)
// per-lane: LDS[lds_off] = MEM[gptr] (16 bytes), tracked by ASYNCcnt.
__device__ inline void async_load_b128(uint32_t lds_off, const void* gptr) {
    asm volatile("global_load_async_to_lds_b128 %0, %1, off"
                 :: "v"(lds_off), "v"((uint64_t)(uintptr_t)gptr)
                 : "memory");
}
__device__ inline void wait_async0() {
    asm volatile("s_wait_asynccnt 0x0" ::: "memory");
}

// ---------------------------------------------------------------- zero init
__global__ void zero_kernel(float4* __restrict__ p, int n4) {
    for (int i = blockIdx.x * blockDim.x + threadIdx.x; i < n4;
         i += gridDim.x * blockDim.x)
        p[i] = make_float4(0.f, 0.f, 0.f, 0.f);
}

// ---------------------------------------------------------------- f32->bf16
__global__ __launch_bounds__(256) void cvt_kernel(
    const float* __restrict__ s, bf16_t* __restrict__ d, int n)
{
    for (int i = blockIdx.x * blockDim.x + threadIdx.x; i < n;
         i += gridDim.x * blockDim.x)
        d[i] = f2bf(s[i]);
}

// ---------------------------------------------------------------- layernorm
// One block per row (256 threads, 4 f32 each), f32 in -> bf16 out.
__global__ __launch_bounds__(256) void ln_kernel(
    const float* __restrict__ in, const float* __restrict__ gw,
    const float* __restrict__ gb, bf16_t* __restrict__ out)
{
    const int row = blockIdx.x;
    const int tid = threadIdx.x;
    const float* x = in + (size_t)row * DIMd;
    float4 xv = ((const float4*)x)[tid];

    float s  = xv.x + xv.y + xv.z + xv.w;
    float ss = xv.x*xv.x + xv.y*xv.y + xv.z*xv.z + xv.w*xv.w;
    #pragma unroll
    for (int o = 16; o; o >>= 1) {
        s  += __shfl_xor(s,  o, 32);
        ss += __shfl_xor(ss, o, 32);
    }
    __shared__ float w1[8], w2[8];
    if ((tid & 31) == 0) { w1[tid >> 5] = s; w2[tid >> 5] = ss; }
    __syncthreads();
    s = 0.f; ss = 0.f;
    #pragma unroll
    for (int i = 0; i < 8; ++i) { s += w1[i]; ss += w2[i]; }

    const float mean = s * (1.0f / DIMd);
    const float var  = ss * (1.0f / DIMd) - mean * mean;
    const float rstd = rsqrtf(var + 1e-5f);

    const int c = tid * 4;
    out[(size_t)row * DIMd + c + 0] = f2bf((xv.x - mean) * rstd * gw[c+0] + gb[c+0]);
    out[(size_t)row * DIMd + c + 1] = f2bf((xv.y - mean) * rstd * gw[c+1] + gb[c+1]);
    out[(size_t)row * DIMd + c + 2] = f2bf((xv.z - mean) * rstd * gw[c+2] + gb[c+2]);
    out[(size_t)row * DIMd + c + 3] = f2bf((xv.w - mean) * rstd * gw[c+3] + gb[c+3]);
}

// ---------------------------------------------------------------- GEMM
// C[M_PAD x Nout] = A[M_PAD x K] * Bw[Nout x K]^T, bf16 in, f32 accum.
// Tile 128x128, BK=32. 256 thr = 8 waves (2 M x 4 N); each wave owns a
// 64x32 sub-tile = 4x2 WMMA accumulators (8 WMMA / K-step, 6 frag loads).
// Double-buffered LDS filled by CDNA5 async-LDS DMA loads (ASYNCcnt):
//   s_wait_asynccnt 0 -> barrier -> issue next tile fills -> 8x WMMA.
// Plus global_prefetch (GL2 warm) two K-steps ahead.
#define BM  128
#define BNt 128
#define BK  32

enum { EPI_QK = 0, EPI_VT = 1, EPI_RESID = 2, EPI_FC1 = 3 };

template<int EPI, int OUTSTRIDE>
__global__ __launch_bounds__(256) void gemm_k(
    const bf16_t* __restrict__ A, const bf16_t* __restrict__ Bw,
    const float* __restrict__ bias, const float* __restrict__ resid,
    float* __restrict__ outF, bf16_t* __restrict__ outB,
    int K, float scale)
{
    __shared__ bf16_t As[2][BM * BK];    // 2 x 8 KB
    __shared__ bf16_t Bs[2][BNt * BK];   // 2 x 8 KB

    const int tid  = threadIdx.x;
    const int lane = tid & 31;
    const int w    = tid >> 5;
    const int wm   = w & 1;          // 0..1 : 64-row slab
    const int wn   = w >> 1;         // 0..3 : 32-col slab
    const int m0   = blockIdx.y * BM;
    const int n0   = blockIdx.x * BNt;

    const int kb  = (lane >> 4) << 4;   // 0 / 16
    const int kb2 = (lane >> 4) << 4;   // 0 / 16

    // cooperative tile-fill assignment: 256 thr x 32 B = 8 KB per operand
    const int lr = tid >> 1;            // 0..127 : row
    const int lh = (tid & 1) * 16;      // 0 / 16 : half-row (in halves)
    const bf16_t* aptr = A  + (size_t)(m0 + lr) * K + lh;
    const bf16_t* bptr = Bw + (size_t)(n0 + lr) * K + lh;
    // AS3 offsets = low 32 bits of the generic pointer (shared aperture is
    // in addr[63:32]); this is the VDST payload async-LDS loads expect.
    const uint32_t aO[2] = { (uint32_t)(uintptr_t)&As[0][lr * BK + lh],
                             (uint32_t)(uintptr_t)&As[1][lr * BK + lh] };
    const uint32_t bO[2] = { (uint32_t)(uintptr_t)&Bs[0][lr * BK + lh],
                             (uint32_t)(uintptr_t)&Bs[1][lr * BK + lh] };

    // prologue: DMA K-step 0 into LDS buffer 0 (2 x 16 B per operand)
    async_load_b128(aO[0],      aptr);
    async_load_b128(aO[0] + 16, aptr + 8);
    async_load_b128(bO[0],      bptr);
    async_load_b128(bO[0] + 16, bptr + 8);

    v8f acc[4][2] = {};
    int p = 0;
    for (int k0 = 0; k0 < K; k0 += BK, p ^= 1) {
        wait_async0();                           // own wave's fills landed
        __syncthreads();                         // all waves' fills landed
        if (k0 + BK < K) {                       // DMA next tile into buf p^1
            const bf16_t* an = aptr + k0 + BK;
            const bf16_t* bn = bptr + k0 + BK;
            async_load_b128(aO[p ^ 1],      an);
            async_load_b128(aO[p ^ 1] + 16, an + 8);
            async_load_b128(bO[p ^ 1],      bn);
            async_load_b128(bO[p ^ 1] + 16, bn + 8);
            if (k0 + 2 * BK < K) {               // warm GL2 two steps ahead
                __builtin_prefetch(aptr + k0 + 2 * BK, 0, 1);
                __builtin_prefetch(bptr + k0 + 2 * BK, 0, 1);
            }
        }

        v16bf af[4], bfr[2];
        #pragma unroll
        for (int i = 0; i < 4; ++i) {
            const int ml = wm * 64 + i * 16 + (lane & 15);
            af[i] = load_afrag(&As[p][ml * BK + kb]);
        }
        #pragma unroll
        for (int j = 0; j < 2; ++j) {
            const int nl = wn * 32 + j * 16 + (lane & 15);
            bfr[j] = *(const v16bf*)&Bs[p][nl * BK + kb2];
        }
        #pragma unroll
        for (int i = 0; i < 4; ++i)
            #pragma unroll
            for (int j = 0; j < 2; ++j)
                acc[i][j] = wmma_bf16(af[i], bfr[j], acc[i][j]);
    }

    // ------------------------------- epilogue (guarded: pad rows never escape)
    #pragma unroll
    for (int i = 0; i < 4; ++i) {
        #pragma unroll
        for (int j = 0; j < 2; ++j) {
            const int mbase = m0 + wm * 64 + i * 16 + ((lane >> 4) << 3);
            const int c     = n0 + wn * 32 + j * 16 + (lane & 15);
            const float bv  = bias[c];
            #pragma unroll
            for (int r = 0; r < 8; ++r) {
                const int m = mbase + r;
                if (m >= M_REAL) continue;
                float val = acc[i][j][r] + bv;
                if (EPI == EPI_QK) {
                    val *= scale;
                    const int bb = m / Nn, nt = m % Nn, hh = c >> 6, d = c & 63;
                    outB[(((size_t)bb * Hh + hh) * QP_PAD + nt) * HDd + d] = f2bf(val);
                } else if (EPI == EPI_VT) {
                    const int bb = m / Nn, nt = m % Nn, hh = c >> 6, d = c & 63;
                    outB[(((size_t)bb * Hh + hh) * HDd + d) * QP_PAD + nt] = f2bf(val);
                } else if (EPI == EPI_RESID) {
                    outF[(size_t)m * OUTSTRIDE + c] =
                        resid[(size_t)m * OUTSTRIDE + c] + val;
                } else { // EPI_FC1: quick_gelu = x * sigmoid(1.702 x)
                    const float g = val / (1.0f + __expf(-1.702f * val));
                    outB[(size_t)m * OUTSTRIDE + c] = f2bf(g);
                }
            }
        }
    }
}

// ---------------------------------------------------------------- attention
// grid (10 qtiles, H, B); block 256 = 8 waves. 32 query rows per block.
// S = Q*K^T (WMMA) -> masked softmax (f32, writes attn_weights) ->
// ctx = P*V (WMMA, V pre-transposed) -> pre_proj (f32) + bf16 ctx buffer.
__global__ __launch_bounds__(256) void attn_kernel(
    const bf16_t* __restrict__ qp, const bf16_t* __restrict__ kp,
    const bf16_t* __restrict__ vT,
    float* __restrict__ attn_out, float* __restrict__ preproj_out,
    bf16_t* __restrict__ ctxb)
{
    __shared__ float  Ssc[32 * QP_PAD];   // 40 KB
    __shared__ bf16_t Pt [32 * QP_PAD];   // 20 KB

    const int qt = blockIdx.x, h = blockIdx.y, b = blockIdx.z;
    const int tid = threadIdx.x, lane = tid & 31, w = tid >> 5;
    const int kb  = (lane >> 4) << 4;
    const int kb2 = (lane >> 4) << 4;

    const bf16_t* qbase = qp + ((size_t)b * Hh + h) * QP_PAD * HDd;
    const bf16_t* kbase = kp + ((size_t)b * Hh + h) * QP_PAD * HDd;
    const bf16_t* vbase = vT + ((size_t)b * Hh + h) * HDd * QP_PAD;

    // ---------------- S = Q*K^T : waves 2(M) x 4(col groups of 80)
    {
        const int wm = w & 1, wn = w >> 1;
        const int am = qt * 32 + wm * 16 + (lane & 15);
        const v16bf aq0 = load_afrag(qbase + (size_t)am * HDd + 0  + kb);
        const v16bf aq1 = load_afrag(qbase + (size_t)am * HDd + 32 + kb);
        #pragma unroll
        for (int nf = 0; nf < 5; ++nf) {
            const int j0 = wn * 80 + nf * 16;
            const int bn = j0 + (lane & 15);
            const v16bf bk0 = *(const v16bf*)(kbase + (size_t)bn * HDd + 0  + kb2);
            const v16bf bk1 = *(const v16bf*)(kbase + (size_t)bn * HDd + 32 + kb2);
            v8f acc = {};
            acc = wmma_bf16(aq0, bk0, acc);
            acc = wmma_bf16(aq1, bk1, acc);
            const int srow0 = wm * 16 + ((lane >> 4) << 3);
            const int scol  = j0 + (lane & 15);
            #pragma unroll
            for (int r = 0; r < 8; ++r) Ssc[(srow0 + r) * QP_PAD + scol] = acc[r];
        }
    }
    __syncthreads();

    // ---------------- softmax over valid columns (j < 257), 8 lanes / row
    {
        const int row = tid >> 3, sub = tid & 7;
        const int qrow = qt * 32 + row;
        float mx = -3.0e38f;
        for (int j = sub; j < Nn; j += 8) mx = fmaxf(mx, Ssc[row * QP_PAD + j]);
        #pragma unroll
        for (int o = 4; o; o >>= 1) mx = fmaxf(mx, __shfl_xor(mx, o, 8));
        float sum = 0.f;
        for (int j = sub; j < Nn; j += 8) sum += __expf(Ssc[row * QP_PAD + j] - mx);
        #pragma unroll
        for (int o = 4; o; o >>= 1) sum += __shfl_xor(sum, o, 8);
        const float inv = 1.0f / sum;
        for (int j = sub; j < QP_PAD; j += 8) {
            const float p = (j < Nn) ? __expf(Ssc[row * QP_PAD + j] - mx) * inv : 0.0f;
            Pt[row * QP_PAD + j] = f2bf(p);
            if (j < Nn && qrow < Nn)
                attn_out[(((size_t)b * Hh + h) * Nn + qrow) * Nn + j] = p;
        }
    }
    __syncthreads();

    // ---------------- ctx = P * V : 8 waves -> 8 (16x16) output frags
    {
        const int mf = w & 1, nf = w >> 1;             // nf 0..3
        const int arow = mf * 16 + (lane & 15);
        const int dcol = nf * 16 + (lane & 15);
        v8f acc = {};
        #pragma unroll
        for (int k0 = 0; k0 < QP_PAD; k0 += 32) {
            const v16bf a  = load_afrag(&Pt[arow * QP_PAD + k0 + kb]);
            const v16bf bv = *(const v16bf*)(vbase + (size_t)dcol * QP_PAD + k0 + kb2);
            acc = wmma_bf16(a, bv, acc);
        }
        const int crow0 = mf * 16 + ((lane >> 4) << 3);
        #pragma unroll
        for (int r = 0; r < 8; ++r) {
            const int qrow = qt * 32 + crow0 + r;
            if (qrow < Nn) {
                const size_t o = ((size_t)b * Nn + qrow) * DIMd + h * HDd
                               + nf * 16 + (lane & 15);
                preproj_out[o] = acc[r];
                ctxb[o]        = f2bf(acc[r]);
            }
        }
    }
}

// ---------------------------------------------------------------- launcher
extern "C" void kernel_launch(void* const* d_in, const int* in_sizes, int n_in,
                              void* d_out, int out_size, void* d_ws, size_t ws_size,
                              hipStream_t stream) {
    (void)in_sizes; (void)n_in; (void)out_size; (void)ws_size;
    const float* hidden = (const float*)d_in[0];
    const float* ln1_w  = (const float*)d_in[1];
    const float* ln1_b  = (const float*)d_in[2];
    const float* Wq = (const float*)d_in[3];  const float* bq = (const float*)d_in[4];
    const float* Wk = (const float*)d_in[5];  const float* bk = (const float*)d_in[6];
    const float* Wv = (const float*)d_in[7];  const float* bv = (const float*)d_in[8];
    const float* Wo = (const float*)d_in[9];  const float* bo = (const float*)d_in[10];
    const float* ln2_w = (const float*)d_in[11];
    const float* ln2_b = (const float*)d_in[12];
    const float* fc1_w = (const float*)d_in[13]; const float* fc1_b = (const float*)d_in[14];
    const float* fc2_w = (const float*)d_in[15]; const float* fc2_b = (const float*)d_in[16];

    // ---- workspace carve-up (all offsets 256B aligned)
    char* ws = (char*)d_ws;
    size_t off = 0;
    auto take = [&](size_t bytes) { char* p = ws + off; off += (bytes + 255) & ~(size_t)255; return p; };
    bf16_t* x_ln = (bf16_t*)take((size_t)M_PAD * DIMd * 2);                 // 17.0 MB
    bf16_t* qp   = (bf16_t*)take((size_t)Bb * Hh * QP_PAD * HDd * 2);       // 21.0 MB
    bf16_t* kp   = (bf16_t*)take((size_t)Bb * Hh * QP_PAD * HDd * 2);       // 21.0 MB
    bf16_t* vT   = (bf16_t*)take((size_t)Bb * Hh * HDd * QP_PAD * 2);       // 21.0 MB
    bf16_t* ctxb = (bf16_t*)take((size_t)M_PAD * DIMd * 2);                 // 17.0 MB
    float*  hbuf = (float*) take((size_t)M_PAD * DIMd * 4);                 // 34.1 MB
    bf16_t* h_ln = (bf16_t*)take((size_t)M_PAD * DIMd * 2);                 // 17.0 MB
    bf16_t* tbuf = (bf16_t*)take((size_t)M_PAD * MLPm * 2);                 // 68.2 MB
    bf16_t* Wqb  = (bf16_t*)take((size_t)DIMd * DIMd * 2);
    bf16_t* Wkb  = (bf16_t*)take((size_t)DIMd * DIMd * 2);
    bf16_t* Wvb  = (bf16_t*)take((size_t)DIMd * DIMd * 2);
    bf16_t* Wob  = (bf16_t*)take((size_t)DIMd * DIMd * 2);
    bf16_t* W1b  = (bf16_t*)take((size_t)MLPm * DIMd * 2);
    bf16_t* W2b  = (bf16_t*)take((size_t)DIMd * MLPm * 2);

    float* out_pre    = (float*)d_out;
    float* out_hidden = out_pre + (size_t)Bb * Nn * DIMd;
    float* out_attn   = out_hidden + (size_t)Bb * Nn * DIMd;

    // 1) zero q/k/vT padded buffers (contiguous region)
    {
        const int n4 = (int)(((size_t)Bb * Hh * QP_PAD * HDd * 2 * 3) / 16);
        zero_kernel<<<2048, 256, 0, stream>>>((float4*)qp, n4);
    }
    // 2) f32 -> bf16 weight conversion (one-shot streaming; ~50 MB @ 23.3 TB/s)
    cvt_kernel<<<2048, 256, 0, stream>>>(Wq, Wqb, DIMd * DIMd);
    cvt_kernel<<<2048, 256, 0, stream>>>(Wk, Wkb, DIMd * DIMd);
    cvt_kernel<<<2048, 256, 0, stream>>>(Wv, Wvb, DIMd * DIMd);
    cvt_kernel<<<2048, 256, 0, stream>>>(Wo, Wob, DIMd * DIMd);
    cvt_kernel<<<4096, 256, 0, stream>>>(fc1_w, W1b, MLPm * DIMd);
    cvt_kernel<<<4096, 256, 0, stream>>>(fc2_w, W2b, DIMd * MLPm);
    // 3) LN1
    ln_kernel<<<M_REAL, 256, 0, stream>>>(hidden, ln1_w, ln1_b, x_ln);
    // 4) QKV projections (WMMA GEMMs)
    gemm_k<EPI_QK, DIMd><<<dim3(DIMd / BNt, M_PAD / BM), 256, 0, stream>>>(
        x_ln, Wqb, bq, nullptr, nullptr, qp, DIMd, SCALEq);
    gemm_k<EPI_QK, DIMd><<<dim3(DIMd / BNt, M_PAD / BM), 256, 0, stream>>>(
        x_ln, Wkb, bk, nullptr, nullptr, kp, DIMd, 1.0f);
    gemm_k<EPI_VT, DIMd><<<dim3(DIMd / BNt, M_PAD / BM), 256, 0, stream>>>(
        x_ln, Wvb, bv, nullptr, nullptr, vT, DIMd, 1.0f);
    // 5) attention: softmax + attn_weights + pre_proj + bf16 ctx
    attn_kernel<<<dim3(QP_PAD / 32, Hh, Bb), 256, 0, stream>>>(
        qp, kp, vT, out_attn, out_pre, ctxb);
    // 6) output projection + residual -> h
    gemm_k<EPI_RESID, DIMd><<<dim3(DIMd / BNt, M_PAD / BM), 256, 0, stream>>>(
        ctxb, Wob, bo, hidden, hbuf, nullptr, DIMd, 1.0f);
    // 7) LN2
    ln_kernel<<<M_REAL, 256, 0, stream>>>(hbuf, ln2_w, ln2_b, h_ln);
    // 8) FC1 + quickGELU
    gemm_k<EPI_FC1, MLPm><<<dim3(MLPm / BNt, M_PAD / BM), 256, 0, stream>>>(
        h_ln, W1b, fc1_b, nullptr, nullptr, tbuf, DIMd, 1.0f);
    // 9) FC2 + residual -> hidden_out
    gemm_k<EPI_RESID, DIMd><<<dim3(DIMd / BNt, M_PAD / BM), 256, 0, stream>>>(
        tbuf, W2b, fc2_b, hbuf, out_hidden, nullptr, MLPm, 1.0f);
}
